// attn_35038343201046
// MI455X (gfx1250) — compile-verified, matrix operand order checked by reading, not run on hardware
//
#include <hip/hip_runtime.h>
#include <hip/hip_bf16.h>

typedef __attribute__((ext_vector_type(16))) __bf16 bf16x16;
typedef __attribute__((ext_vector_type(8)))  __bf16 bf16x8;
typedef __attribute__((ext_vector_type(8)))  float  f32x8;

#define SEQ   2048
#define DIM   4096
#define NH    32
#define NKV   8
#define HD    128
#define KVDIM (NKV * HD)       // 1024
#define RSCALE 0.08838834764831845f   // 1/sqrt(128)

// LDS tile geometry: rows padded 64B -> 80B so the 16 lanes of a b128 fragment
// read start on distinct banks (r*20 mod 64 is a permutation of 16 values).
#define LDS_ROW    80u
#define LDS_ABYTES (128u * LDS_ROW)             // 10240 : A tile 128x32
#define LDS_BUF    (2u * LDS_ABYTES)            // 20480 : A + B (128x32 each)
#define LDS_TOTAL  (2u * LDS_BUF)               // 40960 double-buffered

// ---------------------------------------------------------------------------
// Fragment loader (global): 16x32 bf16 tile (rows x K) from row-major memory.
// CDNA5 16-bit A-layout (ISA 7.12.2): lane = half*16 + r, elements are
// K = [8*half .. 8*half+7] then [16+8*half .. +7]  -> two b128 loads.
// The B-fragment of our NT GEMMs is the identical gather over the [N,K] operand.
// ---------------------------------------------------------------------------
__device__ __forceinline__ bf16x16 load_frag(const __bf16* __restrict__ base,
                                             int ld, int row0, int k0) {
  int lane = threadIdx.x & 31;
  int half = lane >> 4;
  int r    = lane & 15;
  const __bf16* p = base + (size_t)(row0 + r) * ld + k0 + 8 * half;
  union { bf16x16 v; bf16x8 h[2]; } f;
  f.h[0] = *(const bf16x8*)(p);
  f.h[1] = *(const bf16x8*)(p + 16);
  return f.v;
}

// Same fragment gather from a padded LDS tile (row stride LDS_ROW bytes).
__device__ __forceinline__ bf16x16 load_frag_lds(const char* smem,
                                                 unsigned base, int row0) {
  int lane = threadIdx.x & 31;
  unsigned off = base + (unsigned)(row0 + (lane & 15)) * LDS_ROW +
                 (unsigned)((lane >> 4) * 16);
  union { bf16x16 v; bf16x8 h[2]; } f;
  f.h[0] = *(const bf16x8*)(smem + off);
  f.h[1] = *(const bf16x8*)(smem + off + 32);
  return f.v;
}

__device__ __forceinline__ f32x8 wmma_bf16(bf16x16 a, bf16x16 b, f32x8 c) {
  return __builtin_amdgcn_wmma_f32_16x16x32_bf16(
      /*neg_a=*/false, a, /*neg_b=*/false, b,
      /*c_mod=*/(short)0, c, /*reuse_a=*/false, /*reuse_b=*/false);
}

__device__ __forceinline__ void async_b128(unsigned lds, const __bf16* g) {
  asm volatile("global_load_async_to_lds_b128 %0, %1, off"
               :: "v"(lds), "v"(g) : "memory");
}

// ---------------------------------------------------------------------------
// f32 -> bf16 elementwise cast
// ---------------------------------------------------------------------------
__global__ __launch_bounds__(256) void f32_to_bf16_kernel(
    const float* __restrict__ s, __bf16* __restrict__ d, size_t n) {
  size_t i      = (size_t)blockIdx.x * blockDim.x + threadIdx.x;
  size_t stride = (size_t)gridDim.x * blockDim.x;
  for (; i < n; i += stride) d[i] = (__bf16)s[i];
}

// ---------------------------------------------------------------------------
// C[M,N] (f32) = A[M,K] (bf16 row-major) * B[N,K]^T (bf16 row-major)
// block = 256 threads = 8 waves; block tile 128x128; wave tile 32x64
// (2x4 WMMA). A/B 128x32 K-tiles double-buffered in LDS via
// GLOBAL_LOAD_ASYNC_TO_LDS_B128 (ASYNCcnt): each thread owns exactly 4 static
// 16B chunks (2 A rows + 2 B rows), so staging is 4 asm issues, no loop.
// ---------------------------------------------------------------------------
__global__ __launch_bounds__(256) void gemm_bf16_nt(
    const __bf16* __restrict__ A, const __bf16* __restrict__ B,
    float* __restrict__ C, int M, int N, int K) {
  extern __shared__ char smem[];
  // Generic LDS pointer low 32 bits == LDS byte offset (aperture rule).
  unsigned lds_base = (unsigned)(size_t)(void*)smem;

  int wid  = threadIdx.x >> 5;
  int lane = threadIdx.x & 31;
  int wr = wid & 3;        // 4 row groups of 32
  int wc = wid >> 2;       // 2 col groups of 64
  int m_blk = blockIdx.y * 128;
  int n_blk = blockIdx.x * 128;

  // ---- per-thread staging addresses (loop-invariant) ----
  int trow = threadIdx.x >> 2;        // 0..63
  int tcc  = threadIdx.x & 3;         // 16B chunk within a 64B row
  const __bf16* ga0 = A + (size_t)(m_blk + trow) * K + tcc * 8;
  const __bf16* ga1 = ga0 + (size_t)64 * K;
  const __bf16* gb0 = B + (size_t)(n_blk + trow) * K + tcc * 8;
  const __bf16* gb1 = gb0 + (size_t)64 * K;
  unsigned la0 = lds_base + (unsigned)trow * LDS_ROW + (unsigned)tcc * 16u;
  unsigned la1 = la0 + 64u * LDS_ROW;
  unsigned lb0 = la0 + LDS_ABYTES;
  unsigned lb1 = lb0 + 64u * LDS_ROW;

  auto stage = [&](int buf, int k0) {
    unsigned bo = (unsigned)buf * LDS_BUF;
    async_b128(la0 + bo, ga0 + k0);
    async_b128(la1 + bo, ga1 + k0);
    async_b128(lb0 + bo, gb0 + k0);
    async_b128(lb1 + bo, gb1 + k0);
  };

  unsigned aoff = (unsigned)(wr * 32) * LDS_ROW;
  unsigned boff = LDS_ABYTES + (unsigned)(wc * 64) * LDS_ROW;

  f32x8 acc[2][4] = {};

  auto compute = [&](int buf) {
    unsigned bb = (unsigned)buf * LDS_BUF;
    bf16x16 a0 = load_frag_lds(smem, bb + aoff, 0);
    bf16x16 a1 = load_frag_lds(smem, bb + aoff, 16);
#pragma unroll
    for (int j = 0; j < 4; ++j) {
      bf16x16 b = load_frag_lds(smem, bb + boff, j * 16);
      acc[0][j] = wmma_bf16(a0, b, acc[0][j]);
      acc[1][j] = wmma_bf16(a1, b, acc[1][j]);
    }
  };

  stage(0, 0);
  int buf = 0;
  for (int k0 = 0; k0 < K - 32; k0 += 32) {
    asm volatile("s_wait_asynccnt 0x0" ::: "memory");
    __syncthreads();            // staged buffer visible workgroup-wide
    stage(buf ^ 1, k0 + 32);    // prefetch next K-tile into the other buffer
    compute(buf);               // ds reads consumed by WMMA before next barrier
    buf ^= 1;
  }
  asm volatile("s_wait_asynccnt 0x0" ::: "memory");
  __syncthreads();
  compute(buf);

  int m0 = m_blk + wr * 32;
  int n0 = n_blk + wc * 64;
  int half = lane >> 4;
  int cn   = lane & 15;
#pragma unroll
  for (int i = 0; i < 2; ++i)
#pragma unroll
    for (int j = 0; j < 4; ++j)
#pragma unroll
      for (int r = 0; r < 8; ++r)
        C[(size_t)(m0 + 16 * i + 8 * half + r) * N + n0 + 16 * j + cn] =
            acc[i][j][r];
}

// ---------------------------------------------------------------------------
// RoPE on q:  q_f32 [SEQ][NH*HD]  ->  q_bf16 [NH][SEQ][HD], pre-scaled
// ---------------------------------------------------------------------------
__global__ __launch_bounds__(256) void rope_q_kernel(
    const float* __restrict__ qf, __bf16* __restrict__ qb) {
  size_t idx = (size_t)blockIdx.x * blockDim.x + threadIdx.x;
  if (idx >= (size_t)SEQ * NH * (HD / 2)) return;
  int i = idx & 63;          // rotary pair index
  int h = (idx >> 6) & 31;
  int t = (int)(idx >> 11);
  float freq = __expf(-(float)i * 0.14391156f);   // log(1e4)/64
  float ang = (float)t * freq;
  float sn = __sinf(ang), cs = __cosf(ang);
  float xe = qf[(size_t)t * DIM + h * HD + 2 * i];
  float xo = qf[(size_t)t * DIM + h * HD + 2 * i + 1];
  float e =  xe * cs + xo * sn;
  float o = -xe * sn + xo * cs;
  __bf16* dst = qb + ((size_t)h * SEQ + t) * HD;
  dst[2 * i]     = (__bf16)(e * RSCALE);
  dst[2 * i + 1] = (__bf16)(o * RSCALE);
}

// ---------------------------------------------------------------------------
// RoPE on k: k_f32 [SEQ][KVDIM] -> k_out f32 [NH][SEQ][HD] (GQA-repeated,
// part of the reference return) + k_bf16 [NKV][SEQ][HD]
// ---------------------------------------------------------------------------
__global__ __launch_bounds__(256) void rope_k_kernel(
    const float* __restrict__ kf, float* __restrict__ k_out,
    __bf16* __restrict__ kb) {
  size_t idx = (size_t)blockIdx.x * blockDim.x + threadIdx.x;
  if (idx >= (size_t)SEQ * NKV * (HD / 2)) return;
  int i   = idx & 63;
  int kvh = (idx >> 6) & 7;
  int t   = (int)(idx >> 9);
  float freq = __expf(-(float)i * 0.14391156f);
  float ang = (float)t * freq;
  float sn = __sinf(ang), cs = __cosf(ang);
  float xe = kf[(size_t)t * KVDIM + kvh * HD + 2 * i];
  float xo = kf[(size_t)t * KVDIM + kvh * HD + 2 * i + 1];
  float e =  xe * cs + xo * sn;
  float o = -xe * sn + xo * cs;
  for (int rep = 0; rep < 4; ++rep) {
    float* dst = k_out + ((size_t)(kvh * 4 + rep) * SEQ + t) * HD;
    dst[2 * i] = e; dst[2 * i + 1] = o;
  }
  __bf16* db = kb + ((size_t)kvh * SEQ + t) * HD;
  db[2 * i] = (__bf16)e; db[2 * i + 1] = (__bf16)o;
}

// ---------------------------------------------------------------------------
// v: v_f32 [SEQ][KVDIM] -> v_out f32 [NH][SEQ][HD] (repeated) +
//    v^T bf16 [NKV][HD][SEQ]  (transposed for the P·V WMMA A-operand)
// ---------------------------------------------------------------------------
__global__ __launch_bounds__(256) void expand_v_kernel(
    const float* __restrict__ vf, float* __restrict__ v_out,
    __bf16* __restrict__ vtb) {
  size_t idx = (size_t)blockIdx.x * blockDim.x + threadIdx.x;
  if (idx >= (size_t)SEQ * NKV * HD) return;
  int d   = idx & 127;
  int kvh = (idx >> 7) & 7;
  int t   = (int)(idx >> 10);
  float val = vf[(size_t)t * KVDIM + kvh * HD + d];
  for (int rep = 0; rep < 4; ++rep)
    v_out[((size_t)(kvh * 4 + rep) * SEQ + t) * HD + d] = val;
  vtb[((size_t)kvh * HD + d) * SEQ + t] = (__bf16)val;
}

// ---------------------------------------------------------------------------
// Flash attention, one wave per (head, 16-query tile).
//   S^T = K_tile(16x128) x Q^T      (4 chained WMMA per 16-key subtile)
//   online softmax over rows of S^T (in-lane over 8 VGPRs + shfl_xor 16)
//   O^T += V^T(16x32) x P^T         (P^T C-layout -> B-frag is in-lane repack)
// ---------------------------------------------------------------------------
__global__ __launch_bounds__(32) void flash_attn_kernel(
    const __bf16* __restrict__ qb,    // [NH][SEQ][HD]   (pre-scaled, roped)
    const __bf16* __restrict__ kb,    // [NKV][SEQ][HD]  (roped)
    const __bf16* __restrict__ vtb,   // [NKV][HD][SEQ]
    __bf16* __restrict__ attn_out) {  // [SEQ][DIM]
  int q0  = blockIdx.x * 16;
  int h   = blockIdx.y;
  int kvh = h >> 2;
  const __bf16* qh = qb  + (size_t)h   * SEQ * HD;
  const __bf16* kh = kb  + (size_t)kvh * SEQ * HD;
  const __bf16* vh = vtb + (size_t)kvh * HD * SEQ;

  int lane = threadIdx.x & 31;
  int half = lane >> 4;
  int qcol = q0 + (lane & 15);

  bf16x16 qf[4];
  for (int kk = 0; kk < 4; ++kk) qf[kk] = load_frag(qh, HD, q0, kk * 32);

  f32x8 acc[8] = {};
  float m_run = -1e30f, l_run = 0.0f;

  int nkb = (q0 + 15) / 32 + 1;   // causal: skip fully-masked key blocks
  for (int kbk = 0; kbk < nkb; ++kbk) {
    int kbase = kbk * 32;
    f32x8 s0 = {}, s1 = {};
    for (int kk = 0; kk < 4; ++kk) {
      bf16x16 k0f = load_frag(kh, HD, kbase,      kk * 32);
      bf16x16 k1f = load_frag(kh, HD, kbase + 16, kk * 32);
      s0 = wmma_bf16(k0f, qf[kk], s0);
      s1 = wmma_bf16(k1f, qf[kk], s1);
    }
    // causal mask (-1e9 like the reference) + block row-max
    float mb = -1e30f;
    for (int r = 0; r < 8; ++r) {
      int key0 = kbase + 8 * half + r;   // s0 key row
      if (key0 > qcol)      s0[r] += -1.0e9f;
      if (key0 + 16 > qcol) s1[r] += -1.0e9f;
      mb = fmaxf(mb, fmaxf(s0[r], s1[r]));
    }
    mb = fmaxf(mb, __shfl_xor(mb, 16));
    float m_new = fmaxf(m_run, mb);
    float scale = __expf(m_run - m_new);
    float p0[8], p1[8], rs = 0.0f;
    for (int r = 0; r < 8; ++r) {
      p0[r] = __expf(s0[r] - m_new);
      p1[r] = __expf(s1[r] - m_new);
      rs += p0[r] + p1[r];
    }
    rs += __shfl_xor(rs, 16);
    l_run = l_run * scale + rs;
    m_run = m_new;

    // P^T (C-layout, two 16x16 chunks) -> B-fragment: pure in-lane repack
    union { bf16x16 v; __bf16 e[16]; } pb;
    for (int r = 0; r < 8; ++r) {
      pb.e[r]     = (__bf16)p0[r];
      pb.e[8 + r] = (__bf16)p1[r];
    }
    for (int dt = 0; dt < 8; ++dt) {
      bf16x16 vf = load_frag(vh, SEQ, dt * 16, kbase);
      acc[dt] = acc[dt] * scale;
      acc[dt] = wmma_bf16(vf, pb.v, acc[dt]);
    }
  }

  float inv = 1.0f / l_run;
  for (int dt = 0; dt < 8; ++dt)
    for (int r = 0; r < 8; ++r)
      attn_out[(size_t)qcol * DIM + h * HD + dt * 16 + 8 * half + r] =
          (__bf16)(acc[dt][r] * inv);
}

// ---------------------------------------------------------------------------
extern "C" void kernel_launch(void* const* d_in, const int* in_sizes, int n_in,
                              void* d_out, int out_size, void* d_ws,
                              size_t ws_size, hipStream_t stream) {
  const float* x  = (const float*)d_in[0];
  // d_in[1] = mask (causal -1e9 triu) — reproduced analytically in-kernel
  const float* wq = (const float*)d_in[2];
  const float* wk = (const float*)d_in[3];
  const float* wv = (const float*)d_in[4];
  const float* wo = (const float*)d_in[5];

  float* out   = (float*)d_out;                       // [SEQ][DIM]
  float* k_out = out   + (size_t)SEQ * DIM;           // [NH][SEQ][HD]
  float* v_out = k_out + (size_t)SEQ * DIM;           // [NH][SEQ][HD]

  char* p = (char*)d_ws;
  auto carve = [&](size_t bytes) {
    char* r = p;
    p += (bytes + 255) & ~(size_t)255;
    return r;
  };
  __bf16* xb    = (__bf16*)carve((size_t)SEQ * DIM * 2);
  __bf16* wqb   = (__bf16*)carve((size_t)DIM * DIM * 2);
  __bf16* wkb   = (__bf16*)carve((size_t)KVDIM * DIM * 2);
  __bf16* wvb   = (__bf16*)carve((size_t)KVDIM * DIM * 2);
  __bf16* wob   = (__bf16*)carve((size_t)DIM * DIM * 2);
  float*  qf    = (float*) carve((size_t)SEQ * DIM * 4);
  float*  kf    = (float*) carve((size_t)SEQ * KVDIM * 4);
  float*  vf    = (float*) carve((size_t)SEQ * KVDIM * 4);
  __bf16* qb    = (__bf16*)carve((size_t)SEQ * DIM * 2);     // [NH][SEQ][HD]
  __bf16* kb16  = (__bf16*)carve((size_t)SEQ * KVDIM * 2);   // [NKV][SEQ][HD]
  __bf16* vtb   = (__bf16*)carve((size_t)SEQ * KVDIM * 2);   // [NKV][HD][SEQ]
  __bf16* attnb = (__bf16*)carve((size_t)SEQ * DIM * 2);     // [SEQ][DIM]

  // 1) cast operands to bf16
  f32_to_bf16_kernel<<<2048, 256, 0, stream>>>(x,  xb,  (size_t)SEQ * DIM);
  f32_to_bf16_kernel<<<2048, 256, 0, stream>>>(wq, wqb, (size_t)DIM * DIM);
  f32_to_bf16_kernel<<<2048, 256, 0, stream>>>(wk, wkb, (size_t)KVDIM * DIM);
  f32_to_bf16_kernel<<<2048, 256, 0, stream>>>(wv, wvb, (size_t)KVDIM * DIM);
  f32_to_bf16_kernel<<<2048, 256, 0, stream>>>(wo, wob, (size_t)DIM * DIM);

  // 2) q/k/v projections (WMMA + async-LDS double buffering)
  gemm_bf16_nt<<<dim3(DIM / 128,   SEQ / 128), 256, LDS_TOTAL, stream>>>(
      xb, wqb, qf, SEQ, DIM,   DIM);
  gemm_bf16_nt<<<dim3(KVDIM / 128, SEQ / 128), 256, LDS_TOTAL, stream>>>(
      xb, wkb, kf, SEQ, KVDIM, DIM);
  gemm_bf16_nt<<<dim3(KVDIM / 128, SEQ / 128), 256, LDS_TOTAL, stream>>>(
      xb, wvb, vf, SEQ, KVDIM, DIM);

  // 3) rope + GQA expand (also produces the reference's k,v outputs)
  rope_q_kernel<<<(SEQ * NH * (HD / 2)) / 256, 256, 0, stream>>>(qf, qb);
  rope_k_kernel<<<(SEQ * NKV * (HD / 2)) / 256, 256, 0, stream>>>(kf, k_out, kb16);
  expand_v_kernel<<<(SEQ * NKV * HD) / 256, 256, 0, stream>>>(vf, v_out, vtb);

  // 4) flash attention (WMMA)
  flash_attn_kernel<<<dim3(SEQ / 16, NH), 32, 0, stream>>>(qb, kb16, vtb, attnb);

  // 5) output projection (WMMA + async-LDS double buffering)
  gemm_bf16_nt<<<dim3(DIM / 128, SEQ / 128), 256, LDS_TOTAL, stream>>>(
      attnb, wob, out, SEQ, DIM, DIM);
}
